// myVMLSTMCell_NEO3_33517924778468
// MI455X (gfx1250) — matrix operand order repaired and checked by reading, not compiled, not run on hardware
//
#include <hip/hip_runtime.h>
#include <hip/hip_bf16.h>
#include <math.h>

// Problem constants
#define BATCH 16384
#define INF   768
#define HID   1024
#define RANK  64

typedef __attribute__((ext_vector_type(16))) __bf16 v16bf;
typedef __attribute__((ext_vector_type(8)))  __bf16 v8bf;
typedef __attribute__((ext_vector_type(8)))  float  v8f;
typedef __attribute__((ext_vector_type(4)))  unsigned int u32x4;
typedef __attribute__((ext_vector_type(8)))  int          i32x8;

// ---------------- fragment loaders (CDNA5 wave32 WMMA layouts) ----------------
// A matrix 16x32 bf16 (MxK), row-major source with leading dim ld (K-contiguous):
// lanes 0-15: M=lane, K = 0..7 and 16..23 ; lanes 16-31: M=lane-16, K = 8..15 and 24..31
__device__ __forceinline__ v16bf load_fragA(const __bf16* __restrict__ base,
                                            int row, int ld, int kbase, int lane) {
    int m   = lane & 15;
    int sel = lane >> 4;
    const __bf16* p = base + (size_t)(row + m) * ld + kbase + sel * 8;
    v8bf lo = *(const v8bf*)(p);
    v8bf hi = *(const v8bf*)(p + 16);
    return __builtin_shufflevector(lo, hi, 0,1,2,3,4,5,6,7,8,9,10,11,12,13,14,15);
}

// B matrix 32x16 bf16 (KxN), source stored N-major with K contiguous (row stride ld):
// lanes 0-15: N=lane, K = 0..7 and 16..23 ; lanes 16-31: N=lane-16, K = 8..15 and 24..31
__device__ __forceinline__ v16bf load_fragB(const __bf16* base,
                                            int ld, int kbase, int lane) {
    int n   = lane & 15;
    int sel = lane >> 4;
    const __bf16* p = base + (size_t)n * ld + kbase + sel * 8;
    v8bf lo = *(const v8bf*)(p);
    v8bf hi = *(const v8bf*)(p + 16);
    return __builtin_shufflevector(lo, hi, 0,1,2,3,4,5,6,7,8,9,10,11,12,13,14,15);
}

__device__ __forceinline__ float sigmoidf_(float z) {
    return 1.0f / (1.0f + __expf(-z));
}

// ---------------- TDM: 1-D linear copy global -> LDS ---------------------------
// Builds a D# (CDNA5 ISA) for a contiguous `bytes` region
// (8-byte units) and issues tensor_load_to_lds. Call from one wave per block,
// with blockIdx-uniform arguments (descriptor must live in SGPRs).
__device__ __forceinline__ void tdm_copy_1d(void* lds_dst, const void* gsrc, unsigned bytes) {
    unsigned long long ga = (unsigned long long)(size_t)gsrc;
    unsigned lds_off = (unsigned)(size_t)lds_dst;      // generic LDS addr: low 32 bits = offset
    unsigned n8 = bytes >> 3;                          // 8-byte elements

    u32x4 g0;
    g0[0] = 1u;                                        // count=1, user descriptor
    g0[1] = lds_off;                                   // lds_addr [63:32]
    g0[2] = (unsigned)(ga & 0xFFFFFFFFu);              // global_addr [95:64]
    g0[3] = (unsigned)((ga >> 32) & 0x01FFFFFFu)       // global_addr [120:96]
          | 0x80000000u;                               // type=2 ("image")

    i32x8 g1;
    g1[0] = 0x00030000;                                // data_size=3 (8B), no barrier/iter/pad
    g1[1] = (int)((n8 & 0xFFFFu) << 16);               // tensor_dim0[15:0]   (bits 63:48)
    g1[2] = (int)(((n8 >> 16) & 0xFFFFu)               // tensor_dim0[31:16]  (bits 79:64)
          | (1u << 16));                               // tensor_dim1 = 1     (bits 95:80)
    g1[3] = (int)((n8 & 0xFFFFu) << 16);               // tile_dim0           (bits 127:112)
    g1[4] = 1;                                         // tile_dim1 = 1       (bits 143:128)
    g1[5] = (int)n8;                                   // tensor_dim0_stride  (bits 191:160)
    g1[6] = 0;
    g1[7] = 0;

    asm volatile("tensor_load_to_lds %0, %1" :: "s"(g0), "s"(g1) : "memory");
}

// ---------------- stage 1: fp32 -> bf16 bulk convert ----------------
__global__ __launch_bounds__(256) void cvt_f32_bf16(const float* __restrict__ src,
                                                    __bf16* __restrict__ dst, int n) {
    int i0 = (blockIdx.x * 256 + threadIdx.x) * 4;
#pragma unroll
    for (int j = 0; j < 4; ++j) {
        int i = i0 + j;
        if (i < n) dst[i] = (__bf16)src[i];
    }
}

// ---------------- stage 2a: pack Wu/Uu factors, transposed, gate-concat ----------
// dst[n][k], n in [0,256): gate g = n>>6, rank r = n&63 ; src_g is [K][64] row-major
__global__ __launch_bounds__(256) void pack_projT(const float* __restrict__ g0,
                                                  const float* __restrict__ g1,
                                                  const float* __restrict__ g2,
                                                  const float* __restrict__ g3,
                                                  __bf16* __restrict__ dst, int K) {
    int idx = blockIdx.x * 256 + threadIdx.x;
    if (idx >= 256 * K) return;
    int n = idx / K;
    int k = idx - n * K;
    int g = n >> 6;
    int r = n & 63;
    const float* s = (g == 0) ? g0 : (g == 1) ? g1 : (g == 2) ? g2 : g3;
    dst[idx] = (__bf16)s[k * RANK + r];
}

// ---------------- stage 2b: pack fused second factors Bt[n][g][128] -------------
// Layout chosen so each 16-column block tile (4 gates x 16 n x 128 k) is one
// contiguous 16KB region -> single TDM descriptor per block.
// k<64 -> W_g[k][n] ; k>=64 -> U_g[k-64][n]   (W_g,U_g are [64][1024] row-major)
__global__ __launch_bounds__(256) void pack_gateT(const float* __restrict__ Wf, const float* __restrict__ Wi,
                                                  const float* __restrict__ Wc, const float* __restrict__ Wo,
                                                  const float* __restrict__ Uf, const float* __restrict__ Ui,
                                                  const float* __restrict__ Uc, const float* __restrict__ Uo,
                                                  __bf16* __restrict__ dst) {
    int idx = blockIdx.x * 256 + threadIdx.x;      // 1024*4*128 total, dst[(n*4+g)*128+k]
    int n   = idx >> 9;
    int rem = idx & 511;
    int g   = rem >> 7;
    int k   = rem & 127;
    const float* W = (g == 0) ? Wf : (g == 1) ? Wi : (g == 2) ? Wc : Wo;
    const float* U = (g == 0) ? Uf : (g == 1) ? Ui : (g == 2) ? Uc : Uo;
    float v = (k < 64) ? W[k * HID + n] : U[(k - 64) * HID + n];
    dst[idx] = (__bf16)v;
}

// ---------------- stage 3: fold diagonal corrections into coefficients ----------
__global__ __launch_bounds__(256) void coefx_k(const float* __restrict__ a0, const float* __restrict__ a1,
                                               const float* __restrict__ a2, const float* __restrict__ a3,
                                               const float* __restrict__ w0, const float* __restrict__ w1,
                                               const float* __restrict__ w2, const float* __restrict__ w3,
                                               const float* __restrict__ dia, float* __restrict__ out) {
    int idx = blockIdx.x * 256 + threadIdx.x;      // 4*768
    if (idx >= 4 * INF) return;
    int g = idx / INF;
    int i = idx - g * INF;
    const float* a = (g == 0) ? a0 : (g == 1) ? a1 : (g == 2) ? a2 : a3;
    const float* w = (g == 0) ? w0 : (g == 1) ? w1 : (g == 2) ? w2 : w3;
    float dot = 0.f;
#pragma unroll 8
    for (int r = 0; r < RANK; ++r) dot += a[i * RANK + r] * w[r * HID + i];
    out[idx] = dia[i] - dot;
}

__global__ __launch_bounds__(256) void coefh_k(const float* __restrict__ a0, const float* __restrict__ a1,
                                               const float* __restrict__ a2, const float* __restrict__ a3,
                                               const float* __restrict__ u0, const float* __restrict__ u1,
                                               const float* __restrict__ u2, const float* __restrict__ u3,
                                               const float* __restrict__ dia, float* __restrict__ out) {
    int idx = blockIdx.x * 256 + threadIdx.x;      // 4*1024
    if (idx >= 4 * HID) return;
    int g = idx / HID;
    int j = idx - g * HID;
    const float* a = (g == 0) ? a0 : (g == 1) ? a1 : (g == 2) ? a2 : a3;
    const float* u = (g == 0) ? u0 : (g == 1) ? u1 : (g == 2) ? u2 : u3;
    float dot = 0.f;
#pragma unroll 8
    for (int r = 0; r < RANK; ++r) dot += a[j * RANK + r] * u[r * HID + j];
    out[idx] = dia[j] - dot;
}

// ---------------- stage 4: projection GEMM -> PA (bf16, rank-64 per gate) -------
// Block: 8 waves, macro-tile 128(M) x 16(N). The 16xK B tile (rows nTile..nTile+15
// of BtT[256][K], contiguous) is TDM-copied to LDS once; waves read B via ds_load.
__global__ __launch_bounds__(256) void proj_gemm(const __bf16* __restrict__ A,
                                                 const __bf16* __restrict__ Bt,
                                                 __bf16* __restrict__ PA,
                                                 int K, int colbase) {
    __shared__ __align__(16) __bf16 sB[16 * HID];      // up to 32KB (K<=1024)

    int lane  = threadIdx.x & 31;
    int wave  = threadIdx.x >> 5;
    int mTile = blockIdx.x * 128 + wave * 16;
    int nTile = blockIdx.y * 16;

    if (wave == 0) {
        tdm_copy_1d(sB, Bt + (size_t)nTile * K, (unsigned)(16 * K * 2));
        __builtin_amdgcn_s_wait_tensorcnt(0);
    }
    __syncthreads();

    v8f acc = {};
    for (int kk = 0; kk < K; kk += 32) {
        if (kk + 32 < K)
            __builtin_prefetch(A + (size_t)(mTile + (lane & 15)) * K + kk + 32, 0, 1);
        v16bf a = load_fragA(A, mTile, K, kk, lane);
        v16bf b = load_fragB(sB, K, kk, lane);
        acc = __builtin_amdgcn_wmma_f32_16x16x32_bf16(false, a, false, b,
                                                      (short)0, acc, false, false);
    }
    int n       = nTile + (lane & 15);
    int out_col = ((n >> 6) << 7) + colbase + (n & 63);
    int rbase   = mTile + (lane >> 4) * 8;
#pragma unroll
    for (int r = 0; r < 8; ++r)
        PA[(size_t)(rbase + r) * 512 + out_col] = (__bf16)acc[r];
}

// ---------------- stage 5: fused 4-gate GEMM + LSTM epilogue --------------------
// Weight tile Bt[(nTile..nTile+15)*4 + g][0..128] (16KB, contiguous) staged in LDS
// via TDM. pre_g = PA_row_g . Btile_g + bias_g + coef_h*h + coef_x*x, then
// f,i,g,o activations, c' = f*c + i*g, h' = o*tanh(c').
__global__ __launch_bounds__(256) void gate_fused(const __bf16* __restrict__ PA,
                                                  const __bf16* __restrict__ Bt,
                                                  const float* __restrict__ x,
                                                  const float* __restrict__ h,
                                                  const float* __restrict__ c,
                                                  const float* __restrict__ bf_,
                                                  const float* __restrict__ bi_,
                                                  const float* __restrict__ bc_,
                                                  const float* __restrict__ bo_,
                                                  const float* __restrict__ coefx,
                                                  const float* __restrict__ coefh,
                                                  float* __restrict__ out) {
    __shared__ __align__(16) __bf16 sB[64 * 128];      // 16KB: [nLocal][g][k]

    int lane  = threadIdx.x & 31;
    int wave  = threadIdx.x >> 5;
    int mTile = blockIdx.x * 128 + wave * 16;
    int nTile = blockIdx.y * 16;

    if (wave == 0) {
        tdm_copy_1d(sB, Bt + (size_t)nTile * 4 * 128, 64 * 128 * 2);
        __builtin_amdgcn_s_wait_tensorcnt(0);
    }
    __syncthreads();

    v8f acc[4];
#pragma unroll
    for (int g = 0; g < 4; ++g) {
        v8f a = {};
#pragma unroll
        for (int kk = 0; kk < 128; kk += 32) {
            v16bf af = load_fragA(PA, mTile, 512, g * 128 + kk, lane);
            v16bf bf = load_fragB(sB + g * 128, 4 * 128, kk, lane);  // row stride 512
            a = __builtin_amdgcn_wmma_f32_16x16x32_bf16(false, af, false, bf,
                                                        (short)0, a, false, false);
        }
        acc[g] = a;
    }

    int col   = nTile + (lane & 15);
    int rbase = mTile + (lane >> 4) * 8;
    bool inx  = (col < INF);

    float bias[4] = { bf_[col], bi_[col], bc_[col], bo_[col] };
    float chf[4], cxf[4];
#pragma unroll
    for (int g = 0; g < 4; ++g) {
        chf[g] = coefh[g * HID + col];
        cxf[g] = inx ? coefx[g * INF + col] : 0.0f;
    }

#pragma unroll
    for (int r = 0; r < 8; ++r) {
        int row  = rbase + r;
        float hv = h[(size_t)row * HID + col];
        float cv = c[(size_t)row * HID + col];
        float xv = inx ? x[(size_t)row * INF + col] : 0.0f;
        float pre[4];
#pragma unroll
        for (int g = 0; g < 4; ++g)
            pre[g] = acc[g][r] + bias[g] + chf[g] * hv + cxf[g] * xv;
        float fg = sigmoidf_(pre[0]);
        float ig = sigmoidf_(pre[1]);
        float gg = tanhf(pre[2]);
        float og = sigmoidf_(pre[3]);
        float cn = fg * cv + ig * gg;
        float hn = og * tanhf(cn);
        out[(size_t)row * HID + col] = hn;                                  // h_next
        out[(size_t)BATCH * HID + (size_t)row * HID + col] = cn;            // c_next
    }
}

// ---------------- host launch ---------------------------------------------------
extern "C" void kernel_launch(void* const* d_in, const int* in_sizes, int n_in,
                              void* d_out, int out_size, void* d_ws, size_t ws_size,
                              hipStream_t stream) {
    const float* x    = (const float*)d_in[0];
    const float* h    = (const float*)d_in[1];
    const float* c    = (const float*)d_in[2];
    const float* Wu_f = (const float*)d_in[3];
    const float* Wu_i = (const float*)d_in[4];
    const float* Wu_c = (const float*)d_in[5];
    const float* Wu_o = (const float*)d_in[6];
    const float* Uu_f = (const float*)d_in[7];
    const float* Uu_i = (const float*)d_in[8];
    const float* Uu_c = (const float*)d_in[9];
    const float* Uu_o = (const float*)d_in[10];
    const float* Wf   = (const float*)d_in[11];
    const float* Wi   = (const float*)d_in[12];
    const float* Wc   = (const float*)d_in[13];
    const float* Wo   = (const float*)d_in[14];
    const float* Uf   = (const float*)d_in[15];
    const float* Ui   = (const float*)d_in[16];
    const float* Uc   = (const float*)d_in[17];
    const float* Uo   = (const float*)d_in[18];
    const float* b_f  = (const float*)d_in[19];
    const float* b_i  = (const float*)d_in[20];
    const float* b_c  = (const float*)d_in[21];
    const float* b_o  = (const float*)d_in[22];
    const float* Wdia = (const float*)d_in[23];
    const float* Udia = (const float*)d_in[24];

    char* ws = (char*)d_ws;
    const size_t o_xb   = 0;                                  // 16384*768  bf16
    const size_t o_hb   = o_xb   + (size_t)BATCH * INF * 2;   // 16384*1024 bf16
    const size_t o_PA   = o_hb   + (size_t)BATCH * HID * 2;   // 16384*512  bf16
    const size_t o_WubT = o_PA   + (size_t)BATCH * 512 * 2;   // 256*768    bf16
    const size_t o_UubT = o_WubT + (size_t)256 * INF * 2;     // 256*1024   bf16
    const size_t o_Bt   = o_UubT + (size_t)256 * HID * 2;     // 1024*4*128 bf16
    const size_t o_cx   = o_Bt   + (size_t)4 * HID * 128 * 2; // 4*768      f32
    const size_t o_ch   = o_cx   + (size_t)4 * INF * 4;       // 4*1024     f32

    __bf16* xb   = (__bf16*)(ws + o_xb);
    __bf16* hb   = (__bf16*)(ws + o_hb);
    __bf16* PA   = (__bf16*)(ws + o_PA);
    __bf16* WubT = (__bf16*)(ws + o_WubT);
    __bf16* UubT = (__bf16*)(ws + o_UubT);
    __bf16* Bt   = (__bf16*)(ws + o_Bt);
    float*  cx   = (float*)(ws + o_cx);
    float*  ch   = (float*)(ws + o_ch);

    // 1) bf16 copies of the streaming activations
    cvt_f32_bf16<<<(BATCH * INF / 4 + 255) / 256, 256, 0, stream>>>(x, xb, BATCH * INF);
    cvt_f32_bf16<<<(BATCH * HID / 4 + 255) / 256, 256, 0, stream>>>(h, hb, BATCH * HID);

    // 2) weight packing (K-contiguous transposed bf16)
    pack_projT<<<(256 * INF + 255) / 256, 256, 0, stream>>>(Wu_f, Wu_i, Wu_c, Wu_o, WubT, INF);
    pack_projT<<<(256 * HID + 255) / 256, 256, 0, stream>>>(Uu_f, Uu_i, Uu_c, Uu_o, UubT, HID);
    pack_gateT<<<(4 * HID * 128) / 256, 256, 0, stream>>>(Wf, Wi, Wc, Wo, Uf, Ui, Uc, Uo, Bt);

    // 3) fold diagonal corrections
    coefx_k<<<(4 * INF + 255) / 256, 256, 0, stream>>>(Wu_f, Wu_i, Wu_c, Wu_o,
                                                       Wf, Wi, Wc, Wo, Wdia, cx);
    coefh_k<<<(4 * HID + 255) / 256, 256, 0, stream>>>(Uu_f, Uu_i, Uu_c, Uu_o,
                                                       Uf, Ui, Uc, Uo, Udia, ch);

    // 4) rank-64 projections (WMMA): PA[b][g*128+0..63] = x@Wu_g ; [..64..127] = h@Uu_g
    proj_gemm<<<dim3(BATCH / 128, 256 / 16), 256, 0, stream>>>(xb, WubT, PA, INF, 0);
    proj_gemm<<<dim3(BATCH / 128, 256 / 16), 256, 0, stream>>>(hb, UubT, PA, HID, 64);

    // 5) fused 4-gate GEMM + LSTM epilogue
    gate_fused<<<dim3(BATCH / 128, HID / 16), 256, 0, stream>>>(
        PA, Bt, x, h, c, b_f, b_i, b_c, b_o, cx, ch, (float*)d_out);
}
